// Top2Gating_16149077033066
// MI455X (gfx1250) — matrix operand order repaired and checked by reading, not compiled, not run on hardware
//
#include <hip/hip_runtime.h>
#include <hip/hip_bf16.h>

#define TOKENS 8192
#define NEXP   64
#define CAP    256
#define NTILE  (TOKENS / 16)   // 512 tiles of 16 tokens

typedef __attribute__((ext_vector_type(16))) _Float16 v16h;
typedef __attribute__((ext_vector_type(8)))  float    v8f;
typedef __attribute__((ext_vector_type(4)))  float    v4f;

// ---------------------------------------------------------------------------
// K0: zero-fill the 1.07GB output with 128-bit non-temporal stores.
// ---------------------------------------------------------------------------
__global__ void k_fill(float* __restrict__ out, long long n4, long long total) {
    long long i = (long long)blockIdx.x * blockDim.x + threadIdx.x;
    if (i < n4) {
        v4f z = {0.f, 0.f, 0.f, 0.f};
        __builtin_nontemporal_store(z, ((v4f*)out) + i);
    }
    if (i == 0) {  // tail (total = 4*n4 + 1)
        for (long long j = n4 * 4; j < total; ++j) out[j] = 0.f;
    }
}

// ---------------------------------------------------------------------------
// K1: per-token gating. One wave32 per token, 2 experts per lane.
// ---------------------------------------------------------------------------
__global__ void k_gate(const float* __restrict__ logits, const float* __restrict__ noise,
                       int* __restrict__ idx1, int* __restrict__ idx2,
                       float* __restrict__ g1o, float* __restrict__ g2o,
                       float* __restrict__ tmax, float* __restrict__ tinv) {
    int token = blockIdx.x * 8 + (threadIdx.x >> 5);
    int lane  = threadIdx.x & 31;
    const float* row  = logits + token * NEXP;
    const float* nrow = noise  + token * NEXP;
    float l0 = row[lane], l1 = row[lane + 32];

    // row max
    float m = fmaxf(l0, l1);
    for (int off = 16; off; off >>= 1) m = fmaxf(m, __shfl_xor(m, off));
    // softmax denominator
    float ssum = expf(l0 - m) + expf(l1 - m);
    for (int off = 16; off; off >>= 1) ssum += __shfl_xor(ssum, off);
    float inv = 1.0f / ssum;

    // argmax #1 (lowest index on tie)
    float v; int ix;
    if (l1 > l0) { v = l1; ix = lane + 32; } else { v = l0; ix = lane; }
    for (int off = 16; off; off >>= 1) {
        float ov = __shfl_xor(v, off); int oi = __shfl_xor(ix, off);
        if (ov > v || (ov == v && oi < ix)) { v = ov; ix = oi; }
    }
    int   i1    = ix;
    float gate1 = expf(v - m) * inv;   // v == m

    // argmax #2 over (logits + noise) with expert i1 masked to f32 min
    const float NEGMIN = -3.402823466e38f;
    float w0 = (lane        == i1) ? NEGMIN : l0 + nrow[lane];
    float w1 = (lane + 32   == i1) ? NEGMIN : l1 + nrow[lane + 32];
    float v2; int ix2;
    if (w1 > w0) { v2 = w1; ix2 = lane + 32; } else { v2 = w0; ix2 = lane; }
    for (int off = 16; off; off >>= 1) {
        float ov = __shfl_xor(v2, off); int oi = __shfl_xor(ix2, off);
        if (ov > v2 || (ov == v2 && oi < ix2)) { v2 = ov; ix2 = oi; }
    }
    int   i2   = ix2;                      // wave-uniform
    float srcl = (i2 & 32) ? l1 : l0;      // uniform selector
    float li2  = __shfl(srcl, i2 & 31);
    float gate2 = expf(li2 - m) * inv;

    if (lane == 0) {
        idx1[token] = i1; idx2[token] = i2;
        g1o[token] = gate1; g2o[token] = gate2;
        tmax[token] = m; tinv[token] = inv;
    }
}

// ---------------------------------------------------------------------------
// K2: deterministic per-expert sum of gates (for me[e] in l_aux).
// ---------------------------------------------------------------------------
__global__ void k_colsum(const float* __restrict__ logits,
                         const float* __restrict__ tmax, const float* __restrict__ tinv,
                         float* __restrict__ gatesum) {
    int e = blockIdx.x, t = threadIdx.x;   // 64 blocks x 256 threads
    float acc = 0.f;
    for (int s = t; s < TOKENS; s += 256)
        acc += expf(logits[s * NEXP + e] - tmax[s]) * tinv[s];
    __shared__ float red[256];
    red[t] = acc; __syncthreads();
    for (int w = 128; w; w >>= 1) { if (t < w) red[t] += red[t + w]; __syncthreads(); }
    if (t == 0) gatesum[e] = red[0];
}

// ---------------------------------------------------------------------------
// K3: per-16-token-tile inclusive prefix of one-hot masks via WMMA:
//     P(16x16) = L(16x16 lower-tri ones) @ M(16x16 one-hot), f16 in / f32 out.
//     Per-wave tile; 4 expert-groups x 2 masks = 8 v_wmma_f32_16x16x32_f16.
// ---------------------------------------------------------------------------
__global__ void k_tilescan(const int* __restrict__ idx1, const int* __restrict__ idx2,
                           int* __restrict__ r1, int* __restrict__ r2,
                           int* __restrict__ t1, int* __restrict__ t2) {
    __shared__ int   sIdx[2][128];
    __shared__ float sP[8][16][NEXP];
    int tid = threadIdx.x;            // 256 threads = 8 waves, 1 tile per wave
    int wv = tid >> 5, lane = tid & 31;
    int blockTok = blockIdx.x * 128;
    if (tid < 128) sIdx[0][tid]       = idx1[blockTok + tid];
    else           sIdx[1][tid - 128] = idx2[blockTok + tid - 128];
    __syncthreads();

    // A operand: lower-triangular ones L, 16x32 f16 (K=16..31 zero pad).
    // A-layout (contiguous k): lane holds row m=lane&15; element h -> K=(lane<16?0:16)+h.
    int mrow = lane & 15;
    v16h aL;
#pragma unroll
    for (int h = 0; h < 16; ++h) {
        int k = ((lane < 16) ? 0 : 16) + h;
        aL[h] = (k <= mrow) ? (_Float16)1.0f : (_Float16)0.0f;  // k>=16 -> 0
    }

    const _Float16 hone  = (_Float16)1.0f;
    const _Float16 hzero = (_Float16)0.0f;
    int lowmask = (lane < 16) ? 1 : 0;    // lanes >=16 hold K=16..31 (zero pad)
    int ncol    = lane & 15;
    int tile    = blockIdx.x * 8 + wv;

#pragma unroll
    for (int pass = 0; pass < 2; ++pass) {
        const int* sI = &sIdx[pass][wv * 16];
        // Hoist the 16 token->expert ids into registers: unconditional,
        // consecutive LDS reads (merge to b128), single dscnt wait.
        int tk[16];
#pragma unroll
        for (int h = 0; h < 16; ++h) tk[h] = sI[h];

#pragma unroll
        for (int eg = 0; eg < 4; ++eg) {
            int expert = eg * 16 + ncol;
            v16h bM;
#pragma unroll
            for (int h = 0; h < 16; ++h) {
                int match = (tk[h] == expert) & lowmask;   // bitwise: no branches
                bM[h] = match ? hone : hzero;
            }
            v8f c = {};
            v8f d = __builtin_amdgcn_wmma_f32_16x16x32_f16(
                false, aL, false, bM, (short)0, c, false, false);
            int mbase = (lane < 16) ? 0 : 8;
#pragma unroll
            for (int r = 0; r < 8; ++r) sP[wv][mbase + r][expert] = d[r];
        }
        __syncthreads();
        int* ro = pass ? r2 : r1;
        int* to = pass ? t2 : t1;
        if (lane < 16) {   // per-token inclusive rank within tile (1..16)
            int tok = wv * 16 + lane;
            int e = sI[lane];
            ro[blockTok + tok] = (int)sP[wv][lane][e];
        }
        // tile totals (pre-drop per-expert histogram) = row 15 of P
        to[tile * NEXP + lane]      = (int)sP[wv][15][lane];
        to[tile * NEXP + lane + 32] = (int)sP[wv][15][lane + 32];
        __syncthreads();
    }
}

// ---------------------------------------------------------------------------
// K4: exclusive scan of tile totals per expert + l_aux.
//     off1 folds the -1; off2 folds (-1 + total count1[e]).
// ---------------------------------------------------------------------------
__global__ void k_scan(const int* __restrict__ t1, const int* __restrict__ t2,
                       const float* __restrict__ gatesum,
                       int* __restrict__ off1, int* __restrict__ off2,
                       float* __restrict__ outL) {
    int e = threadIdx.x;   // 1 block x 64 threads
    int run = 0;
    for (int i = 0; i < NTILE; ++i) {
        int v = t1[i * NEXP + e];
        off1[i * NEXP + e] = run - 1;
        run += v;
    }
    int total1 = run;            // pre-drop sum(mask1)[e]
    int run2 = total1 - 1;
    for (int i = 0; i < NTILE; ++i) {
        int v = t2[i * NEXP + e];
        off2[i * NEXP + e] = run2;
        run2 += v;
    }
    __shared__ float red[64];
    red[e] = gatesum[e] * (float)total1;
    __syncthreads();
    for (int w = 32; w; w >>= 1) { if (e < w) red[e] += red[e + w]; __syncthreads(); }
    if (e == 0)
        outL[0] = (float)NEXP * red[0] / ((float)TOKENS * (float)TOKENS);
}

// ---------------------------------------------------------------------------
// K5: capacity drop, post-drop renormalization, sparse scatter into d_out.
// ---------------------------------------------------------------------------
__global__ void k_scatter(const int* __restrict__ idx1, const int* __restrict__ idx2,
                          const float* __restrict__ g1, const float* __restrict__ g2,
                          const int* __restrict__ r1, const int* __restrict__ r2,
                          const int* __restrict__ off1, const int* __restrict__ off2,
                          float* __restrict__ out) {
    int s = blockIdx.x * 256 + threadIdx.x;
    if (s >= TOKENS) return;
    int tile = s >> 4;
    int e1 = idx1[s], e2 = idx2[s];
    int loc1 = off1[tile * NEXP + e1] + r1[s];
    int loc2 = off2[tile * NEXP + e2] + r2[s];
    bool k1 = loc1 < CAP;
    bool k2 = loc2 < CAP;
    float a = k1 ? g1[s] : 0.f;
    float b = k2 ? g2[s] : 0.f;
    float denom = fmaxf(a + b, 1.1920929e-07f);   // finfo(f32).eps
    float w1 = a / denom, w2 = b / denom;
    float* cw = out + 1;
    float* dm = out + 1 + (long long)TOKENS * NEXP * CAP;
    if (k1) {
        long long o = ((long long)s * NEXP + e1) * CAP + loc1;
        cw[o] = w1; dm[o] = 1.0f;
    }
    if (k2) {
        long long o = ((long long)s * NEXP + e2) * CAP + loc2;
        cw[o] = w2; dm[o] = 1.0f;
    }
}

// ---------------------------------------------------------------------------
extern "C" void kernel_launch(void* const* d_in, const int* in_sizes, int n_in,
                              void* d_out, int out_size, void* d_ws, size_t ws_size,
                              hipStream_t stream) {
    const float* logits = (const float*)d_in[0];
    const float* noise  = (const float*)d_in[1];
    float* out = (float*)d_out;

    // workspace layout (4-byte words)
    int*   w     = (int*)d_ws;
    int*   idx1  = w;                         // 8192
    int*   idx2  = idx1 + TOKENS;             // 8192
    float* g1    = (float*)(idx2 + TOKENS);   // 8192
    float* g2    = g1 + TOKENS;               // 8192
    float* tmax  = g2 + TOKENS;               // 8192
    float* tinv  = tmax + TOKENS;             // 8192
    int*   r1    = (int*)(tinv + TOKENS);     // 8192
    int*   r2    = r1 + TOKENS;               // 8192
    int*   t1    = r2 + TOKENS;               // 512*64
    int*   t2    = t1 + NTILE * NEXP;         // 512*64
    int*   off1  = t2 + NTILE * NEXP;         // 512*64
    int*   off2  = off1 + NTILE * NEXP;       // 512*64
    float* gsum  = (float*)(off2 + NTILE * NEXP); // 64

    long long total = 1LL + 2LL * TOKENS * NEXP * CAP;   // 268,435,457 floats
    long long n4    = total / 4;                          // 67,108,864 float4s

    k_fill<<<(unsigned)((n4 + 255) / 256), 256, 0, stream>>>(out, n4, total);
    k_gate<<<TOKENS / 8, 256, 0, stream>>>(logits, noise, idx1, idx2, g1, g2, tmax, tinv);
    k_colsum<<<NEXP, 256, 0, stream>>>(logits, tmax, tinv, gsum);
    k_tilescan<<<NTILE / 8, 256, 0, stream>>>(idx1, idx2, r1, r2, t1, t2);
    k_scan<<<1, NEXP, 0, stream>>>(t1, t2, gsum, off1, off2, out);
    k_scatter<<<(TOKENS + 255) / 256, 256, 0, stream>>>(idx1, idx2, g1, g2, r1, r2,
                                                        off1, off2, out);
}
